// SwinTransformerBlock_35175782154428
// MI455X (gfx1250) — compile-verified, matrix operand order checked by reading, not run on hardware
//
#include <hip/hip_runtime.h>

// ---------------------------------------------------------------------------
// Fused Swin transformer block for gfx1250 (MI455X).
// One block = one 8x8 shifted window (64 tokens, C=64, 4 heads x hd=16).
// 4096 blocks x 128 threads (4 wave32). All GEMMs via v_wmma_f32_16x16x32_f16.
// ---------------------------------------------------------------------------

typedef __attribute__((ext_vector_type(16))) _Float16 v16h;
typedef __attribute__((ext_vector_type(8)))  _Float16 h8;   // 16B -> ds_load_b128
typedef __attribute__((ext_vector_type(8)))  float    v8f;

#define WMMA_F16(a, b, c) \
  __builtin_amdgcn_wmma_f32_16x16x32_f16(false, (a), false, (b), (short)0, (c), false, false)

__device__ __forceinline__ v8f vzero8() {
  v8f z = {0.f, 0.f, 0.f, 0.f, 0.f, 0.f, 0.f, 0.f};
  return z;
}

// A-style fragment (16x32 f16) from an LDS row-major tile, ld in halves.
// lane L: row = L%16 ; K-runs {hi..hi+7} and {hi+16..hi+23}, hi = 8*(L/16).
// Two 16B loads per lane -> ds_load_b128.
__device__ __forceinline__ v16h frag_lds(const _Float16* p0, int ld, int lane) {
  const _Float16* p = p0 + (lane & 15) * ld + ((lane >> 4) << 3);
  union { v16h v; h8 h[2]; } u;
  u.h[0] = *(const h8*)(p);
  u.h[1] = *(const h8*)(p + 16);
  return u.v;
}

// B-style fragment built from a row-major f32 weight matrix W[out][in] in
// global memory (B[k][n] = W[n][k]): lane L owns out-row nBase + L%16 and
// reads two contiguous 8-float runs, converting f32->f16 in-register.
// Weights are <=32KB and reused by all 4096 blocks -> L2 resident.
__device__ __forceinline__ v16h frag_w32(const float* __restrict__ W, int ld,
                                         int nBase, int kBase, int lane) {
  const float* p = W + (size_t)(nBase + (lane & 15)) * ld + kBase + ((lane >> 4) << 3);
  v16h v;
#pragma unroll
  for (int j = 0; j < 8; ++j) v[j] = (_Float16)p[j];
#pragma unroll
  for (int j = 0; j < 8; ++j) v[8 + j] = (_Float16)p[16 + j];
  return v;
}

// ---------------- LDS pool (time-multiplexed, 64KB total) -------------------
// phase1 : RAWX f32 ld68 [0,17408) | XN16 ld72 [17408,26624) | Y16 ld72 [26624,35840)
// phase2 : Q16 ld72 [35840,45056) | K16P 4x(64x40) [45056,65536) | V16T ld72 [0,9216)
// phase3 : MASKS f32 ld64 [9216,25600)
// phase4 : P heads ld72 at {9216,35840,45056,54272}
// phase5 : O16 ld64 [18432,26624)
// phase6 : XO f32 ld68 [0,17408) | XH16 ld72 [17408,26624)
// phase7 : H16 ld136 [26624,44032) | FIN f32 ld68 [44544,61952)
// All fragment bases/strides are 16B aligned.
// ---------------------------------------------------------------------------

__global__ __launch_bounds__(128)
void swin_block_fused(const float* __restrict__ x, const float* __restrict__ maskg,
                      const float* __restrict__ rpb, const float* __restrict__ qk_w,
                      const float* __restrict__ qk_b, const float* __restrict__ v_w,
                      const float* __restrict__ v_b, const float* __restrict__ proj_w,
                      const float* __restrict__ proj_b, const float* __restrict__ n1w,
                      const float* __restrict__ n1b, const float* __restrict__ n2w,
                      const float* __restrict__ n2b, const float* __restrict__ fc1_w,
                      const float* __restrict__ fc1_b, const float* __restrict__ fc2_w,
                      const float* __restrict__ fc2_b, float* __restrict__ out) {
  __shared__ __align__(128) char smem[65536];

  float*    rawX = (float*)(smem + 0);         // ld 68
  _Float16* v16T = (_Float16*)(smem + 0);      // ld 72  (V transposed: [d][token])
  float*    maskS = (float*)(smem + 9216);     // ld 64
  _Float16* xn16 = (_Float16*)(smem + 17408);  // ld 72
  _Float16* y16  = (_Float16*)(smem + 26624);  // ld 72  (shortcut, kept live)
  _Float16* q16  = (_Float16*)(smem + 35840);  // ld 72
  _Float16* k16p = (_Float16*)(smem + 45056);  // per-head 64x40, cols>=16 zeroed
  _Float16* O16  = (_Float16*)(smem + 18432);  // ld 64
  float*    xoS  = (float*)(smem + 0);         // ld 68
  _Float16* xh16 = (_Float16*)(smem + 17408);  // ld 72
  _Float16* H16  = (_Float16*)(smem + 26624);  // ld 136
  float*    finS = (float*)(smem + 44544);     // ld 68

  const int tid  = threadIdx.x;
  const int lane = tid & 31;
  const int wv   = tid >> 5;          // wave id 0..3 (= head id in attention)
  const int nlo  = lane & 15;
  const int hb   = (lane >> 4) << 3;  // 0 or 8 (C-layout M offset)

  const int blk = blockIdx.x;
  const int b  = blk >> 10;           // batch
  const int wf = blk & 1023;          // spatial window (for mask)
  const int wy = wf >> 5, wx = wf & 31;
  const int w0 = (wx << 3) + 4;       // shifted base column (roll -4 folded in)

  __builtin_prefetch(qk_w, 0, 1);

  // ---- phase 1: gather shifted window; sx -> rawX (f32), sy -> y16 (f16) ----
#pragma unroll
  for (int it = 0; it < 4; ++it) {
    int idx = it * 128 + tid;              // (c, window-row)
    int c = idx >> 3, r = idx & 7;
    int hh = ((wy << 3) + r + 4) & 255;
    const float* px = x + (((size_t)b * 128 + c) << 16) + ((size_t)hh << 8);
    const float* py = px + ((size_t)64 << 16);
    float vx[8], vy[8];
    if (wx != 31) {
#pragma unroll
      for (int k = 0; k < 8; ++k) { vx[k] = px[w0 + k]; vy[k] = py[w0 + k]; }
    } else {
#pragma unroll
      for (int k = 0; k < 8; ++k) { int ww = (w0 + k) & 255; vx[k] = px[ww]; vy[k] = py[ww]; }
    }
#pragma unroll
    for (int k = 0; k < 8; ++k) {
      int t = r * 8 + k;
      rawX[t * 68 + c] = vx[k];
      y16[t * 72 + c]  = (_Float16)vy[k];
    }
  }
  __syncthreads();

  // ---- phase 2a: LayerNorm1 (threads 0..63) + zero k-pad (threads 64..127) --
  if (tid < 64) {
    const int t = tid;
    float mu = 0.f;
#pragma unroll
    for (int c = 0; c < 64; ++c) mu += rawX[t * 68 + c];
    mu *= (1.f / 64.f);
    float var = 0.f;
#pragma unroll
    for (int c = 0; c < 64; ++c) { float d = rawX[t * 68 + c] - mu; var += d * d; }
    float rs = rsqrtf(var * (1.f / 64.f) + 1e-5f);
#pragma unroll
    for (int c = 0; c < 64; ++c)
      xn16[t * 72 + c] = (_Float16)((rawX[t * 68 + c] - mu) * rs * n1w[c] + n1b[c]);
#pragma unroll
    for (int p = 64; p < 72; ++p) q16[t * 72 + p] = (_Float16)0.f;  // finite K-pad
  } else {
    unsigned* kz = (unsigned*)k16p;  // zero entire padded-K buffer (5120 dwords)
    int t2 = tid - 64;
#pragma unroll
    for (int i = 0; i < 80; ++i) kz[t2 * 80 + i] = 0u;
  }
  __syncthreads();

  // ---- phase 2b: q,k = LN(x) @ qk_w^T + b ; wave w -> q cols 16w.., k head w -
  {
    v8f accQ[4], accK[4];
#pragma unroll
    for (int mi = 0; mi < 4; ++mi) { accQ[mi] = vzero8(); accK[mi] = vzero8(); }
#pragma unroll
    for (int ks = 0; ks < 2; ++ks) {
      v16h bq = frag_w32(qk_w, 64, 16 * wv, 32 * ks, lane);
      v16h bk = frag_w32(qk_w, 64, 64 + 16 * wv, 32 * ks, lane);
#pragma unroll
      for (int mi = 0; mi < 4; ++mi) {
        v16h a = frag_lds(xn16 + mi * (16 * 72) + 32 * ks, 72, lane);
        accQ[mi] = WMMA_F16(a, bq, accQ[mi]);
        accK[mi] = WMMA_F16(a, bk, accK[mi]);
      }
    }
    float qb = qk_b[16 * wv + nlo];
    float kb = qk_b[64 + 16 * wv + nlo];
#pragma unroll
    for (int mi = 0; mi < 4; ++mi)
#pragma unroll
      for (int r = 0; r < 8; ++r) {
        int t = 16 * mi + r + hb;
        q16[t * 72 + 16 * wv + nlo]   = (_Float16)(accQ[mi][r] + qb);
        k16p[wv * 2560 + t * 40 + nlo] = (_Float16)(accK[mi][r] + kb); // K>=16 stays 0
      }
  }
  // ---- phase 2c: v = y @ v_w^T + b, stored TRANSPOSED as v16T[d][token] -----
  {
    v8f accV[4];
#pragma unroll
    for (int mi = 0; mi < 4; ++mi) accV[mi] = vzero8();
#pragma unroll
    for (int ks = 0; ks < 2; ++ks) {
      v16h bv = frag_w32(v_w, 64, 16 * wv, 32 * ks, lane);
#pragma unroll
      for (int mi = 0; mi < 4; ++mi) {
        v16h a = frag_lds(y16 + mi * (16 * 72) + 32 * ks, 72, lane);
        accV[mi] = WMMA_F16(a, bv, accV[mi]);
      }
    }
    float vb = v_b[16 * wv + nlo];
#pragma unroll
    for (int mi = 0; mi < 4; ++mi)
#pragma unroll
      for (int r = 0; r < 8; ++r) {
        int t = 16 * mi + r + hb;
        v16T[(16 * wv + nlo) * 72 + t] = (_Float16)(accV[mi][r] + vb);
      }
  }
  __syncthreads();

  // ---- phase 3: stage this window's attention mask (16KB, coalesced) --------
  {
    const float* mp = maskg + (size_t)wf * 4096;
#pragma unroll
    for (int i = 0; i < 32; ++i) maskS[tid + i * 128] = mp[tid + i * 128];
  }
  __syncthreads();

  // ---- phase 4: scores (K padded 16->32), bias/mask, register softmax, P ----
  {
    v8f S[16];
#pragma unroll
    for (int i = 0; i < 16; ++i) S[i] = vzero8();
    v16h bK[4];
#pragma unroll
    for (int ni = 0; ni < 4; ++ni)
      bK[ni] = frag_lds(k16p + wv * 2560 + ni * (16 * 40), 40, lane);
#pragma unroll
    for (int mi = 0; mi < 4; ++mi) {
      v16h a = frag_lds(q16 + mi * (16 * 72) + 16 * wv, 72, lane);
#pragma unroll
      for (int ni = 0; ni < 4; ++ni) S[mi * 4 + ni] = WMMA_F16(a, bK[ni], S[mi * 4 + ni]);
    }
    // scale + relative position bias + shift mask
#pragma unroll
    for (int mi = 0; mi < 4; ++mi)
#pragma unroll
      for (int ni = 0; ni < 4; ++ni)
#pragma unroll
        for (int r = 0; r < 8; ++r) {
          int t = 16 * mi + r + hb, n = 16 * ni + nlo;
          int rel = ((t >> 3) - (n >> 3) + 7) * 15 + ((t & 7) - (n & 7) + 7);
          S[mi * 4 + ni][r] = S[mi * 4 + ni][r] * 0.25f + maskS[t * 64 + n] + rpb[rel * 4 + wv];
        }
    // softmax: each row lives in one 16-lane half -> shfl_xor 1/2/4/8
#pragma unroll
    for (int mi = 0; mi < 4; ++mi)
#pragma unroll
      for (int r = 0; r < 8; ++r) {
        float m = fmaxf(fmaxf(S[mi * 4][r], S[mi * 4 + 1][r]),
                        fmaxf(S[mi * 4 + 2][r], S[mi * 4 + 3][r]));
        m = fmaxf(m, __shfl_xor(m, 1, 32));
        m = fmaxf(m, __shfl_xor(m, 2, 32));
        m = fmaxf(m, __shfl_xor(m, 4, 32));
        m = fmaxf(m, __shfl_xor(m, 8, 32));
        float s = 0.f;
#pragma unroll
        for (int ni = 0; ni < 4; ++ni) {
          float e = __expf(S[mi * 4 + ni][r] - m);
          S[mi * 4 + ni][r] = e;
          s += e;
        }
        s += __shfl_xor(s, 1, 32);
        s += __shfl_xor(s, 2, 32);
        s += __shfl_xor(s, 4, 32);
        s += __shfl_xor(s, 8, 32);
        float inv = 1.f / s;
#pragma unroll
        for (int ni = 0; ni < 4; ++ni) S[mi * 4 + ni][r] *= inv;
      }
    __syncthreads();  // everyone done reading q16/k16p/maskS
    _Float16* P = (_Float16*)(smem + (wv == 0 ? 9216 : 35840 + (wv - 1) * 9216));
#pragma unroll
    for (int mi = 0; mi < 4; ++mi)
#pragma unroll
      for (int ni = 0; ni < 4; ++ni)
#pragma unroll
        for (int r = 0; r < 8; ++r) {
          int t = 16 * mi + r + hb, n = 16 * ni + nlo;
          P[t * 72 + n] = (_Float16)S[mi * 4 + ni][r];
        }
  }
  __syncthreads();

  // ---- phase 5: O = P @ V (per head); B-frags from v16T rows (already ^T) ---
  {
    const _Float16* P = (const _Float16*)(smem + (wv == 0 ? 9216 : 35840 + (wv - 1) * 9216));
    v8f O[4];
#pragma unroll
    for (int mi = 0; mi < 4; ++mi) O[mi] = vzero8();
#pragma unroll
    for (int ks = 0; ks < 2; ++ks) {
      v16h bV = frag_lds(v16T + (16 * wv) * 72 + 32 * ks, 72, lane);
#pragma unroll
      for (int mi = 0; mi < 4; ++mi) {
        v16h a = frag_lds(P + mi * (16 * 72) + 32 * ks, 72, lane);
        O[mi] = WMMA_F16(a, bV, O[mi]);
      }
    }
#pragma unroll
    for (int mi = 0; mi < 4; ++mi)
#pragma unroll
      for (int r = 0; r < 8; ++r) {
        int t = 16 * mi + r + hb;
        O16[t * 64 + 16 * wv + nlo] = (_Float16)O[mi][r];
      }
  }
  __syncthreads();

  // ---- phase 6: proj + residual(shortcut = sy) -> xo (f32) ------------------
  {
    v8f pr[4];
#pragma unroll
    for (int mi = 0; mi < 4; ++mi) pr[mi] = vzero8();
#pragma unroll
    for (int ks = 0; ks < 2; ++ks) {
      v16h bp = frag_w32(proj_w, 64, 16 * wv, 32 * ks, lane);
#pragma unroll
      for (int mi = 0; mi < 4; ++mi) {
        v16h a = frag_lds(O16 + mi * (16 * 64) + 32 * ks, 64, lane);
        pr[mi] = WMMA_F16(a, bp, pr[mi]);
      }
    }
    float pb = proj_b[16 * wv + nlo];
#pragma unroll
    for (int mi = 0; mi < 4; ++mi)
#pragma unroll
      for (int r = 0; r < 8; ++r) {
        int t = 16 * mi + r + hb, col = 16 * wv + nlo;
        xoS[t * 68 + col] = pr[mi][r] + pb + (float)y16[t * 72 + col];
      }
  }
  __syncthreads();

  // ---- phase 7: LayerNorm2 -> xh16 ------------------------------------------
  if (tid < 64) {
    const int t = tid;
    float mu = 0.f;
#pragma unroll
    for (int c = 0; c < 64; ++c) mu += xoS[t * 68 + c];
    mu *= (1.f / 64.f);
    float var = 0.f;
#pragma unroll
    for (int c = 0; c < 64; ++c) { float d = xoS[t * 68 + c] - mu; var += d * d; }
    float rs = rsqrtf(var * (1.f / 64.f) + 1e-5f);
#pragma unroll
    for (int c = 0; c < 64; ++c)
      xh16[t * 72 + c] = (_Float16)((xoS[t * 68 + c] - mu) * rs * n2w[c] + n2b[c]);
  }
  __syncthreads();

  // ---- phase 8: fc1 (64->128) + exact GELU -> H16 ---------------------------
  {
    v8f f1[8];
#pragma unroll
    for (int i = 0; i < 8; ++i) f1[i] = vzero8();
#pragma unroll
    for (int ks = 0; ks < 2; ++ks) {
      v16h b0 = frag_w32(fc1_w, 64, 32 * wv, 32 * ks, lane);
      v16h b1 = frag_w32(fc1_w, 64, 32 * wv + 16, 32 * ks, lane);
#pragma unroll
      for (int mi = 0; mi < 4; ++mi) {
        v16h a = frag_lds(xh16 + mi * (16 * 72) + 32 * ks, 72, lane);
        f1[mi * 2 + 0] = WMMA_F16(a, b0, f1[mi * 2 + 0]);
        f1[mi * 2 + 1] = WMMA_F16(a, b1, f1[mi * 2 + 1]);
      }
    }
#pragma unroll
    for (int j = 0; j < 2; ++j) {
      float fb = fc1_b[32 * wv + 16 * j + nlo];
#pragma unroll
      for (int mi = 0; mi < 4; ++mi)
#pragma unroll
        for (int r = 0; r < 8; ++r) {
          int t = 16 * mi + r + hb;
          float h = f1[mi * 2 + j][r] + fb;
          float g = 0.5f * h * (1.f + erff(h * 0.70710678118f));
          H16[t * 136 + 32 * wv + 16 * j + nlo] = (_Float16)g;
        }
    }
  }
  __syncthreads();

  // ---- phase 9: fc2 (128->64) + residual(xo) -> final -----------------------
  {
    v8f f2[4];
#pragma unroll
    for (int mi = 0; mi < 4; ++mi) f2[mi] = vzero8();
#pragma unroll
    for (int ks = 0; ks < 4; ++ks) {
      v16h bb = frag_w32(fc2_w, 128, 16 * wv, 32 * ks, lane);
#pragma unroll
      for (int mi = 0; mi < 4; ++mi) {
        v16h a = frag_lds(H16 + mi * (16 * 136) + 32 * ks, 136, lane);
        f2[mi] = WMMA_F16(a, bb, f2[mi]);
      }
    }
    float ob = fc2_b[16 * wv + nlo];
#pragma unroll
    for (int mi = 0; mi < 4; ++mi)
#pragma unroll
      for (int r = 0; r < 8; ++r) {
        int t = 16 * mi + r + hb, col = 16 * wv + nlo;
        finS[t * 68 + col] = f2[mi][r] + ob + xoS[t * 68 + col];
      }
  }
  __syncthreads();

  // ---- phase 10: scatter to (B,C,H,W) with roll(+4,+4) folded in ------------
#pragma unroll
  for (int it = 0; it < 4; ++it) {
    int idx = it * 128 + tid;
    int c = idx >> 3, r = idx & 7;
    int hh = ((wy << 3) + r + 4) & 255;
    float* po = out + (((size_t)b * 64 + c) << 16) + ((size_t)hh << 8);
    if (wx != 31) {
#pragma unroll
      for (int k = 0; k < 8; ++k) po[w0 + k] = finS[(r * 8 + k) * 68 + c];
    } else {
#pragma unroll
      for (int k = 0; k < 8; ++k) po[(w0 + k) & 255] = finS[(r * 8 + k) * 68 + c];
    }
  }
}

extern "C" void kernel_launch(void* const* d_in, const int* in_sizes, int n_in,
                              void* d_out, int out_size, void* d_ws, size_t ws_size,
                              hipStream_t stream) {
  (void)in_sizes; (void)n_in; (void)out_size; (void)d_ws; (void)ws_size;
  const float* x      = (const float*)d_in[0];
  const float* mask   = (const float*)d_in[1];
  const float* rpb    = (const float*)d_in[2];
  const float* qk_w   = (const float*)d_in[3];
  const float* qk_b   = (const float*)d_in[4];
  const float* v_w    = (const float*)d_in[5];
  const float* v_b    = (const float*)d_in[6];
  const float* proj_w = (const float*)d_in[7];
  const float* proj_b = (const float*)d_in[8];
  const float* n1w    = (const float*)d_in[9];
  const float* n1b    = (const float*)d_in[10];
  const float* n2w    = (const float*)d_in[11];
  const float* n2b    = (const float*)d_in[12];
  const float* fc1_w  = (const float*)d_in[13];
  const float* fc1_b  = (const float*)d_in[14];
  const float* fc2_w  = (const float*)d_in[15];
  const float* fc2_b  = (const float*)d_in[16];

  dim3 grid(4096), block(128);
  swin_block_fused<<<grid, block, 0, stream>>>(
      x, mask, rpb, qk_w, qk_b, v_w, v_b, proj_w, proj_b,
      n1w, n1b, n2w, n2b, fc1_w, fc1_b, fc2_w, fc2_b, (float*)d_out);
}
